// multihead_attention_35064113004751
// MI455X (gfx1250) — matrix operand order repaired and checked by reading, not compiled, not run on hardware
//
#include <hip/hip_runtime.h>

#define S_LEN 2048
#define DM    1024
#define NH    16
#define DH    64
#define BATCH 2
#define TOK   (BATCH * S_LEN)   // 4096 tokens
#define XE    (TOK * DM)        // 4,194,304
#define WE    (DM * DM)         // 1,048,576

typedef __bf16 bf16;
typedef __attribute__((ext_vector_type(16))) __bf16 v16bf;
typedef __attribute__((ext_vector_type(8)))  __bf16 v8bf;
typedef __attribute__((ext_vector_type(8)))  float  v8f;

union Frag { v16bf v; v8bf h[2]; };

__device__ __forceinline__ v8f wmma_bf16(const Frag& a, const Frag& b, v8f c) {
  // D(16x16,f32) = A(16x32,bf16) x B(32x16,bf16) + C
  return __builtin_amdgcn_wmma_f32_16x16x32_bf16(false, a.v, false, b.v,
                                                 (short)0, c, false, false);
}

// Async DMA: 16 bytes global -> LDS, tracked by ASYNCcnt (gfx1250).
__device__ __forceinline__ void async_b128(const void* g, void* l) {
  const unsigned lv = (unsigned)(size_t)l;   // low 32 bits of generic LDS addr = LDS offset
  asm volatile("global_load_async_to_lds_b128 %0, %1, off"
               :: "v"(lv), "v"(g) : "memory");
}

// ---------------------------------------------------------------- converts
__global__ void cvt_x_kernel(const float* __restrict__ x, bf16* __restrict__ xb) {
  const size_t i = (size_t)blockIdx.x * blockDim.x + threadIdx.x;
  if (i < (size_t)XE) xb[i] = (bf16)x[i];
}

// W [K_in, N_out] f32  ->  Wt [N_out, K_in] bf16 (B-fragment friendly)
__global__ void cvt_wT_kernel(const float* __restrict__ w0, const float* __restrict__ w1,
                              const float* __restrict__ w2, bf16* __restrict__ t0,
                              bf16* __restrict__ t1, bf16* __restrict__ t2) {
  const size_t i = (size_t)blockIdx.x * blockDim.x + threadIdx.x;
  if (i >= (size_t)WE) return;
  const int k = (int)(i >> 10);
  const int n = (int)(i & (DM - 1));
  const float* w = (blockIdx.y == 0) ? w0 : (blockIdx.y == 1 ? w1 : w2);
  bf16*       wt = (blockIdx.y == 0) ? t0 : (blockIdx.y == 1 ? t1 : t2);
  wt[(size_t)n * DM + k] = (bf16)w[i];
}

// ---------------------------------------------------------------- QKV GEMM
// out = (x @ W + b) * premul, bf16 in [B,H,S,Dh] (vmode=0) or [B,H,Dh,S] (vmode=1)
__global__ __launch_bounds__(256)
void qkv_gemm_kernel(const bf16* __restrict__ xb, const bf16* __restrict__ wt,
                     const float* __restrict__ bias, bf16* __restrict__ out,
                     int vmode, float premul)
{
  const int lane = threadIdx.x & 31;
  const int wid  = threadIdx.x >> 5;
  const int ln16 = lane & 15;
  const int hi   = lane >> 4;
  const int m0 = blockIdx.x * 128 + wid * 16;   // token tile
  const int n0 = blockIdx.y * 64;               // output-feature tile

  v8f zero = {};
  v8f acc[4];
#pragma unroll
  for (int t = 0; t < 4; ++t) acc[t] = zero;

  const bf16* arow = xb + (size_t)(m0 + ln16) * DM + hi * 8;
  for (int k0 = 0; k0 < DM; k0 += 32) {
    Frag a;
    a.h[0] = *(const v8bf*)(arow + k0);
    a.h[1] = *(const v8bf*)(arow + k0 + 16);
#pragma unroll
    for (int t = 0; t < 4; ++t) {
      const bf16* bcol = wt + (size_t)(n0 + t * 16 + ln16) * DM + k0 + hi * 8;
      Frag b;
      b.h[0] = *(const v8bf*)(bcol);
      b.h[1] = *(const v8bf*)(bcol + 16);
      acc[t] = wmma_bf16(a, b, acc[t]);
    }
  }

#pragma unroll
  for (int t = 0; t < 4; ++t) {
    const int n = n0 + t * 16 + ln16;
    const float bv = bias[n];
    const int h = n >> 6, d = n & (DH - 1);
#pragma unroll
    for (int r = 0; r < 8; ++r) {
      const int m = m0 + r + hi * 8;
      const int bidx = m >> 11;
      const int s = m & (S_LEN - 1);
      const float val = (acc[t][r] + bv) * premul;
      size_t idx;
      if (vmode == 0) idx = (((size_t)(bidx * NH + h) * S_LEN) + s) * DH + d;
      else            idx = (((size_t)(bidx * NH + h) * DH) + d) * S_LEN + s;
      out[idx] = (bf16)val;
    }
  }
}

// ---------------------------------------------------------------- attention
// Flash-style streaming softmax. Workgroup = 4 waves = 64 query rows, one (b,h).
// K/V tiles (32 keys) staged in LDS via double-buffered async DMA, shared by all waves.
__global__ __launch_bounds__(128)
void attn_kernel(const bf16* __restrict__ Qb, const bf16* __restrict__ Kb,
                 const bf16* __restrict__ Vt, float* __restrict__ ctx)
{
  __shared__ __bf16 kvbuf[2][4096];      // per buf: [0..2047] K 32x64, [2048..4095] V 64x32
  __shared__ __bf16 plds[4 * 16 * 32];   // per-wave 16x32 P staging tile

  const int tid  = threadIdx.x;
  const int lane = tid & 31;
  const int wid  = tid >> 5;
  const int ln16 = lane & 15;
  const int hi   = lane >> 4;
  const int bh = blockIdx.y;                      // b*16 + h
  const int q0 = blockIdx.x * 64 + wid * 16;

  const bf16* Kbase = Kb + (size_t)bh * S_LEN * DH;   // 32-key tile is contiguous 2048 elems
  const bf16* Vbase = Vt + (size_t)bh * DH * S_LEN;   // [Dh, S]

  // async staging map: 4 x b128 per thread per tile
  const int f1 = tid * 8, f2 = 1024 + tid * 8;
  const int vr1 = f1 >> 5, vc1 = f1 & 31;
  const int vr2 = f2 >> 5, vc2 = f2 & 31;
  auto stage = [&](int k0, __bf16* buf) {
    async_b128(Kbase + (size_t)k0 * DH + f1, buf + f1);
    async_b128(Kbase + (size_t)k0 * DH + f2, buf + f2);
    async_b128(Vbase + (size_t)vr1 * S_LEN + k0 + vc1, buf + 2048 + f1);
    async_b128(Vbase + (size_t)vr2 * S_LEN + k0 + vc2, buf + 2048 + f2);
  };

  // Q fragments for this wave's 16 rows (scale 1/8 pre-folded in GEMM epilogue)
  Frag qa[2];
  {
    const bf16* qp = Qb + ((size_t)bh * S_LEN + q0 + ln16) * DH + hi * 8;
#pragma unroll
    for (int c = 0; c < 2; ++c) {
      qa[c].h[0] = *(const v8bf*)(qp + c * 32);
      qa[c].h[1] = *(const v8bf*)(qp + c * 32 + 16);
    }
  }

  float m8[8], l8[8];
#pragma unroll
  for (int r = 0; r < 8; ++r) { m8[r] = -1e30f; l8[r] = 0.f; }
  v8f zero = {};
  v8f o[4];
#pragma unroll
  for (int t = 0; t < 4; ++t) o[t] = zero;

  __bf16* pl = plds + wid * (16 * 32);
  const int NIT = S_LEN / 32;

  stage(0, kvbuf[0]);                     // prefetch tile 0

  for (int it = 0; it < NIT; ++it) {
    const __bf16* cur = kvbuf[it & 1];
    const int kn = (it + 1 < NIT) ? (it + 1) * 32 : 0;   // dummy refetch on last iter
    stage(kn, kvbuf[(it + 1) & 1]);
    // in-order async completion: <=4 outstanding => current tile's 4 have landed
    asm volatile("s_wait_asynccnt 4" ::: "memory");
    __syncthreads();

    const __bf16* bk = cur;          // K tile [32 keys][64 d]
    const __bf16* bv = cur + 2048;   // V tile [64 d][32 keys]

    // ---- scores from LDS: S0 = keys 0..15, S1 = keys 16..31 of tile
    v8f s0 = zero, s1 = zero;
#pragma unroll
    for (int t = 0; t < 2; ++t) {
      const __bf16* kp = bk + (t * 16 + ln16) * 64;
#pragma unroll
      for (int c = 0; c < 2; ++c) {
        Frag b;
        b.h[0] = *(const v8bf*)(kp + c * 32 + hi * 8);
        b.h[1] = *(const v8bf*)(kp + c * 32 + hi * 8 + 16);
        if (t == 0) s0 = wmma_bf16(qa[c], b, s0);
        else        s1 = wmma_bf16(qa[c], b, s1);
      }
    }

    // ---- online softmax update (row reductions across 16-lane column groups)
    float mx[8];
#pragma unroll
    for (int r = 0; r < 8; ++r) mx[r] = fmaxf(s0[r], s1[r]);
#pragma unroll
    for (int d = 1; d < 16; d <<= 1) {
#pragma unroll
      for (int r = 0; r < 8; ++r) mx[r] = fmaxf(mx[r], __shfl_xor(mx[r], d, 32));
    }
    float p0[8], p1[8], rs[8], corr[8];
#pragma unroll
    for (int r = 0; r < 8; ++r) {
      const float mn = fmaxf(m8[r], mx[r]);
      corr[r] = __expf(m8[r] - mn);
      p0[r] = __expf(s0[r] - mn);
      p1[r] = __expf(s1[r] - mn);
      rs[r] = p0[r] + p1[r];
      m8[r] = mn;
    }
#pragma unroll
    for (int d = 1; d < 16; d <<= 1) {
#pragma unroll
      for (int r = 0; r < 8; ++r) rs[r] += __shfl_xor(rs[r], d, 32);
    }
#pragma unroll
    for (int r = 0; r < 8; ++r) l8[r] = l8[r] * corr[r] + rs[r];
#pragma unroll
    for (int t = 0; t < 4; ++t)
#pragma unroll
      for (int r = 0; r < 8; ++r) o[t][r] *= corr[r];

    // ---- re-layout P: C-fragment (f32) -> LDS (bf16) -> A-fragment
#pragma unroll
    for (int r = 0; r < 8; ++r) {
      const int row = r + hi * 8;
      pl[row * 32 + ln16]      = (__bf16)p0[r];
      pl[row * 32 + 16 + ln16] = (__bf16)p1[r];
    }
    asm volatile("s_wait_dscnt 0" ::: "memory");
    Frag pf;
    const __bf16* pr = pl + ln16 * 32 + hi * 8;
    pf.h[0] = *(const v8bf*)(pr);
    pf.h[1] = *(const v8bf*)(pr + 16);

    // ---- O += P @ V from LDS
#pragma unroll
    for (int t = 0; t < 4; ++t) {
      const __bf16* vp = bv + (t * 16 + ln16) * 32 + hi * 8;
      Frag b;
      b.h[0] = *(const v8bf*)(vp);
      b.h[1] = *(const v8bf*)(vp + 16);
      o[t] = wmma_bf16(pf, b, o[t]);
    }
    __syncthreads();   // all waves done reading this buffer before it is re-staged
  }

  // ---- epilogue: normalize and write merged-heads ctx [B,S,D] f32
  const int b = bh >> 4, h = bh & (NH - 1);
  float inv[8];
#pragma unroll
  for (int r = 0; r < 8; ++r) inv[r] = 1.0f / l8[r];
#pragma unroll
  for (int t = 0; t < 4; ++t) {
#pragma unroll
    for (int r = 0; r < 8; ++r) {
      const int row = r + hi * 8;
      ctx[((size_t)(b * S_LEN + q0 + row)) * DM + h * DH + t * 16 + ln16] =
          o[t][r] * inv[r];
    }
  }
}

// ---------------------------------------------------------------- residual + LayerNorm
__global__ __launch_bounds__(256)
void resid_ln_kernel(const float* __restrict__ ctx, const float* __restrict__ x,
                     const float* __restrict__ gamma, const float* __restrict__ beta,
                     float* __restrict__ out)
{
  __shared__ float s_sum[8], s_sq[8];
  const int row = blockIdx.x;
  const int tid = threadIdx.x;
  const int lane = tid & 31, wid = tid >> 5;
  const size_t base = (size_t)row * DM;
  float v[4], s = 0.f, ss = 0.f;
#pragma unroll
  for (int j = 0; j < 4; ++j) {
    const int i = tid * 4 + j;
    v[j] = ctx[base + i] + x[base + i];
    s += v[j]; ss += v[j] * v[j];
  }
#pragma unroll
  for (int d = 16; d >= 1; d >>= 1) {
    s  += __shfl_xor(s, d, 32);
    ss += __shfl_xor(ss, d, 32);
  }
  if (lane == 0) { s_sum[wid] = s; s_sq[wid] = ss; }
  __syncthreads();
  float S = 0.f, SS = 0.f;
#pragma unroll
  for (int w = 0; w < 8; ++w) { S += s_sum[w]; SS += s_sq[w]; }
  const float mean = S * (1.0f / DM);
  const float var  = SS * (1.0f / DM) - mean * mean;
  const float rstd = rsqrtf(var + 1e-3f);
#pragma unroll
  for (int j = 0; j < 4; ++j) {
    const int i = tid * 4 + j;
    out[base + i] = (v[j] - mean) * rstd * gamma[i] + beta[i];
  }
}

// ---------------------------------------------------------------- launch
extern "C" void kernel_launch(void* const* d_in, const int* in_sizes, int n_in,
                              void* d_out, int out_size, void* d_ws, size_t ws_size,
                              hipStream_t stream)
{
  (void)in_sizes; (void)n_in; (void)out_size; (void)ws_size;
  const float* x  = (const float*)d_in[0];
  const float* Wq = (const float*)d_in[1];
  const float* bq = (const float*)d_in[2];
  const float* Wk = (const float*)d_in[3];
  const float* bk = (const float*)d_in[4];
  const float* Wv = (const float*)d_in[5];
  const float* bv = (const float*)d_in[6];
  const float* gm = (const float*)d_in[7];
  const float* bt = (const float*)d_in[8];
  float* out = (float*)d_out;

  char* ws = (char*)d_ws;
  bf16*  xb  = (bf16*)(ws);                 //  8 MB  x bf16
  bf16*  wtq = (bf16*)(ws + 8388608);       //  2 MB  Wq^T bf16
  bf16*  wtk = (bf16*)(ws + 10485760);      //  2 MB
  bf16*  wtv = (bf16*)(ws + 12582912);      //  2 MB
  bf16*  Qb  = (bf16*)(ws + 14680064);      //  8 MB  [B,H,S,Dh] (pre-scaled by 1/8)
  bf16*  Kb  = (bf16*)(ws + 23068672);      //  8 MB  [B,H,S,Dh]
  bf16*  Vt  = (bf16*)(ws + 31457280);      //  8 MB  [B,H,Dh,S]
  float* ctx = (float*)(ws + 39845888);     // 16 MB  [B,S,D]

  cvt_x_kernel<<<XE / 256, 256, 0, stream>>>(x, xb);
  cvt_wT_kernel<<<dim3(WE / 256, 3), 256, 0, stream>>>(Wq, Wk, Wv, wtq, wtk, wtv);
  qkv_gemm_kernel<<<dim3(TOK / 128, DM / 64), 256, 0, stream>>>(xb, wtq, bq, Qb, 0, 0.125f);
  qkv_gemm_kernel<<<dim3(TOK / 128, DM / 64), 256, 0, stream>>>(xb, wtk, bk, Kb, 0, 1.0f);
  qkv_gemm_kernel<<<dim3(TOK / 128, DM / 64), 256, 0, stream>>>(xb, wtv, bv, Vt, 1, 1.0f);
  attn_kernel<<<dim3(S_LEN / 64, BATCH * NH), 128, 0, stream>>>(Qb, Kb, Vt, ctx);
  resid_ln_kernel<<<TOK, 256, 0, stream>>>(ctx, x, gm, bt, out);
}
